// MHLA_90838558311151
// MI455X (gfx1250) — matrix operand order repaired and checked by reading, not compiled, not run on hardware
//
#include <hip/hip_runtime.h>
#include <hip/hip_bf16.h>

// ---------------------------------------------------------------------------
// Types
// ---------------------------------------------------------------------------
typedef __bf16 bf16_t;
typedef bf16_t bf16x16 __attribute__((ext_vector_type(16)));
typedef float  f32x8   __attribute__((ext_vector_type(8)));
typedef unsigned int u32x4 __attribute__((ext_vector_type(4)));
typedef unsigned int v4u  __attribute__((ext_vector_type(4)));
typedef int          v8i  __attribute__((ext_vector_type(8)));
typedef int          v4i  __attribute__((ext_vector_type(4)));

union FragB16 { bf16x16 v; u32x4 q[2]; unsigned short h[16]; };

__device__ __forceinline__ unsigned short f2bf(float f) {
  unsigned u = __float_as_uint(f);
  unsigned r = u + 0x7fffu + ((u >> 16) & 1u);   // round-to-nearest-even
  return (unsigned short)(r >> 16);
}

__device__ __forceinline__ f32x8 wmma_bf16(const FragB16& a, const FragB16& b, f32x8 c) {
  return __builtin_amdgcn_wmma_f32_16x16x32_bf16(false, a.v, false, b.v,
                                                 (short)0, c, false, false);
}

// ---------------------------------------------------------------------------
// Tensor Data Mover: 2-D bf16 tile load Global -> LDS.
// D# packing per CDNA5 ISA:
//   G0: [1:0]=count=1, [63:32]=lds_addr, [120:64]=global_addr, [127:126]=type=2
//   G1: [17:16]=data_size=1(2B), [79:48]=tensor_dim0, [111:80]=tensor_dim1,
//       [127:112]=tile_dim0, [143:128]=tile_dim1, [207:160]=tensor_dim0_stride
// ---------------------------------------------------------------------------
__device__ __forceinline__ void tdm_load_2d_bf16(unsigned lds_off, const void* gptr,
                                                 unsigned dim0, unsigned dim1,
                                                 unsigned stride0) {
  unsigned long long ga = (unsigned long long)gptr;
  v4u g0;
  g0[0] = 1u;                                   // count=1 (valid), user mode
  g0[1] = lds_off;                              // lds_addr (bytes)
  g0[2] = (unsigned)(ga & 0xffffffffu);         // global_addr[31:0]
  g0[3] = (unsigned)((ga >> 32) & 0x01ffffffu) | (2u << 30);  // ga[56:32] | type=2
  v8i g1;
  g1[0] = (int)(1u << 16);                      // data_size=1 -> 2 bytes
  g1[1] = (int)(dim0 << 16);                    // tensor_dim0[15:0] @ bit48
  g1[2] = (int)((dim0 >> 16) | (dim1 << 16));   // tensor_dim0 hi | tensor_dim1 lo
  g1[3] = (int)((dim1 >> 16) | (dim0 << 16));   // tensor_dim1 hi | tile_dim0 (=dim0)
  g1[4] = (int)(dim1 & 0xffffu);                // tile_dim1 (=dim1), tile_dim2=0
  g1[5] = (int)stride0;                         // tensor_dim0_stride[31:0]
  g1[6] = 0;                                    // stride0 hi16 | dim1_stride lo
  g1[7] = 0;
  v4i z4 = {0, 0, 0, 0};
#if defined(__clang_major__) && (__clang_major__ >= 23)
  v8i z8 = {0, 0, 0, 0, 0, 0, 0, 0};
  __builtin_amdgcn_tensor_load_to_lds(g0, g1, z4, z4, z8, 0);
#else
  __builtin_amdgcn_tensor_load_to_lds(g0, g1, z4, z4, 0);
#endif
}

__device__ __forceinline__ unsigned lds_offset_of(const void* p) {
  return (unsigned)(unsigned long long)(uintptr_t)p;   // low 32 bits = LDS offset
}

// ---------------------------------------------------------------------------
// f32 -> bf16 conversion
// ---------------------------------------------------------------------------
__global__ void k_f32_to_bf16(const float* __restrict__ src,
                              unsigned short* __restrict__ dst, int n) {
  int i = blockIdx.x * blockDim.x + threadIdx.x;
  int stride = gridDim.x * blockDim.x;
  for (; i < n; i += stride) dst[i] = f2bf(src[i]);
}

// ---------------------------------------------------------------------------
// Generic bf16 WMMA GEMM: C[M,N] = A[M,K] * B[K,N], row-major.
// Block tile 64x128, 8 waves (4m x 2n), each wave 16x64, K-step 32.
// A tile arrives via double-buffered TDM; B tile manually transposed into LDS.
// ---------------------------------------------------------------------------
__global__ __launch_bounds__(256)
void k_gemm_bf16(const unsigned short* __restrict__ A,
                 const unsigned short* __restrict__ B,
                 float* __restrict__ Cf, unsigned short* __restrict__ Cb,
                 int M, int N, int K) {
  __shared__ unsigned short At[2][64 * 32];  // [row][k], double-buffered (TDM dst)
  __shared__ unsigned short Bt[128 * 32];    // transposed: [col][k]
  const int tid  = threadIdx.x;
  const int lane = tid & 31;
  const int wid  = tid >> 5;
  const int wm   = wid & 3;
  const int wn   = wid >> 2;
  const int m0 = blockIdx.y * 64;
  const int n0 = blockIdx.x * 128;

  const f32x8 zacc = {0.f,0.f,0.f,0.f,0.f,0.f,0.f,0.f};
  f32x8 acc[4];
  for (int i = 0; i < 4; ++i) acc[i] = zacc;

  const int nk = K >> 5;
  // Prologue: TDM for k-step 0 into buffer 0 (wave 0 issues; TDM ignores EXEC).
  if (wid == 0) {
    tdm_load_2d_bf16(lds_offset_of(&At[0][0]), A + (size_t)m0 * K,
                     32u, 64u, (unsigned)K);
  }

  for (int kk = 0; kk < nk; ++kk) {
    const int k0 = kk << 5;
    const int buf = kk & 1;
    // Issue TDM for next k-step into the other buffer, then wait for current.
    if (wid == 0) {
      if (kk + 1 < nk) {
        tdm_load_2d_bf16(lds_offset_of(&At[buf ^ 1][0]),
                         A + (size_t)m0 * K + (k0 + 32),
                         32u, 64u, (unsigned)K);
        __builtin_amdgcn_s_wait_tensorcnt(1);
      } else {
        __builtin_amdgcn_s_wait_tensorcnt(0);
      }
    }
    // B tile transposed: 32x128 halves = 512 chunks, two per thread
    for (int cc = 0; cc < 2; ++cc) {
      int c = tid + cc * 256;
      int kr = c >> 4, col8 = (c & 15) * 8;
      unsigned short hv[8];
      if (n0 + col8 + 8 <= N) {
        u32x4 d = *(const u32x4*)(B + (size_t)(k0 + kr) * N + n0 + col8);
        const unsigned short* p = (const unsigned short*)&d;
        for (int i = 0; i < 8; ++i) hv[i] = p[i];
      } else {
        for (int i = 0; i < 8; ++i) hv[i] = 0;
      }
      for (int i = 0; i < 8; ++i) Bt[(col8 + i) * 32 + kr] = hv[i];
    }
    __syncthreads();

    // A fragment (16x32 bf16): lanes 0-15 K{0..7,16..23}, lanes 16-31 K{8..15,24..31}
    FragB16 a;
    {
      int row = wm * 16 + (lane & 15);
      int c0 = (lane >> 4) * 8;
      a.q[0] = *(const u32x4*)(&At[buf][0] + row * 32 + c0);
      a.q[1] = *(const u32x4*)(&At[buf][0] + row * 32 + c0 + 16);
    }
    // Load all four B fragments first, then burst the WMMAs back-to-back.
    FragB16 b[4];
    const int kb = (lane >> 4) * 8;
    for (int nt = 0; nt < 4; ++nt) {
      int col = wn * 64 + nt * 16 + (lane & 15);
      b[nt].q[0] = *(const u32x4*)(Bt + col * 32 + kb);
      b[nt].q[1] = *(const u32x4*)(Bt + col * 32 + kb + 16);
    }
    for (int nt = 0; nt < 4; ++nt) acc[nt] = wmma_bf16(a, b[nt], acc[nt]);
    __syncthreads();
  }

  // C/D layout: row = vv + 8*(lane>=16), col = lane&15
  for (int nt = 0; nt < 4; ++nt) {
    int col = n0 + wn * 64 + nt * 16 + (lane & 15);
    if (col < N) {
      for (int vv = 0; vv < 8; ++vv) {
        int row = m0 + wm * 16 + vv + (lane >> 4) * 8;
        float val = acc[nt][vv];
        if (Cf) Cf[(size_t)row * N + col] = val;
        if (Cb) Cb[(size_t)row * N + col] = f2bf(val);
      }
    }
  }
}

// ---------------------------------------------------------------------------
// Prep: RoPE + concat + RMSNorm -> qf/kf [b,h,t,96] bf16, V reorder -> [b,h,t,64]
// One block per (b,t) row; 8 waves x 2 heads.
// ---------------------------------------------------------------------------
__global__ __launch_bounds__(256)
void k_prep(const float* __restrict__ qstate, const float* __restrict__ qrot,
            const float* __restrict__ kstate, const float* __restrict__ krot,
            const float* __restrict__ vstate,
            unsigned short* __restrict__ qf, unsigned short* __restrict__ kf,
            unsigned short* __restrict__ vT, int T) {
  const int r = blockIdx.x;             // b*T + t
  const int bi = r >> 11, ti = r & (T - 1);
  const int lane = threadIdx.x & 31, wid = threadIdx.x >> 5;
  const float kLn1e4_16 = 0.57564627324851148f;  // ln(10000)/16

  for (int hh = 0; hh < 2; ++hh) {
    int h = wid * 2 + hh;
    size_t hdst = (size_t)(bi * 16 + h) * T + ti;

    for (int which = 0; which < 2; ++which) {
      const float* st = which ? kstate : qstate;
      const float* rt = which ? (krot + (size_t)r * 32)
                              : (qrot + (size_t)r * 512 + h * 32);
      unsigned short* dst = which ? kf : qf;
      float vals[3]; float ss = 0.f;
      for (int u = 0; u < 3; ++u) {
        int e = lane * 3 + u;
        float val;
        if (e < 64) {
          val = st[(size_t)r * 1024 + h * 64 + e];
        } else {
          int jj = e - 64;
          int fidx = (jj < 16) ? jj : jj - 16;
          float ang = (float)ti * __expf(-(float)fidx * kLn1e4_16);
          float c = __cosf(ang), s = __sinf(ang);
          val = (jj < 16) ? (rt[jj] * c - rt[jj + 16] * s)
                          : (rt[jj - 16] * s + rt[jj] * c);
        }
        vals[u] = val; ss += val * val;
      }
      for (int off = 16; off >= 1; off >>= 1) ss += __shfl_xor(ss, off, 32);
      float sc = rsqrtf(ss * (1.0f / 96.0f) + 1e-6f);
      for (int u = 0; u < 3; ++u)
        dst[hdst * 96 + lane * 3 + u] = f2bf(vals[u] * sc);
    }
    for (int e = lane; e < 64; e += 32)
      vT[hdst * 64 + e] = f2bf(vstate[(size_t)r * 1024 + h * 64 + e]);
  }
}

// ---------------------------------------------------------------------------
// Flash attention: block = 4 waves, 64 queries; per-wave 16 queries.
// S (16x96 . 96x32) and PV (16x32 . 32x64) via bf16 WMMA, online softmax.
// K tiles via double-buffered TDM; V tiles transposed into LDS manually.
// ---------------------------------------------------------------------------
__global__ __launch_bounds__(128)
void k_attn(const unsigned short* __restrict__ qf,
            const unsigned short* __restrict__ kf,
            const unsigned short* __restrict__ vT,
            unsigned short* __restrict__ attn, int T) {
  __shared__ unsigned short Kt[2][32 * 96]; // [keyrow][k], double-buffered (TDM dst)
  __shared__ unsigned short Vt[64 * 32];    // transposed [n][k]
  __shared__ unsigned short Pt[4][16 * 32];
  const int tid = threadIdx.x, lane = tid & 31, wid = tid >> 5;
  const int m0b = blockIdx.x * 64;
  const int h = blockIdx.y, bi = blockIdx.z;
  const int m0w = m0b + wid * 16;
  const size_t headbase = (size_t)(bi * 16 + h) * T;

  // Q fragments: 3 K-chunks of 32
  FragB16 aq[3];
  {
    const unsigned short* qrow = qf + (headbase + m0w + (lane & 15)) * 96;
    int c0 = (lane >> 4) * 8;
    for (int kc = 0; kc < 3; ++kc) {
      aq[kc].q[0] = *(const u32x4*)(qrow + kc * 32 + c0);
      aq[kc].q[1] = *(const u32x4*)(qrow + kc * 32 + c0 + 16);
    }
  }

  const f32x8 zacc = {0.f,0.f,0.f,0.f,0.f,0.f,0.f,0.f};
  f32x8 o[4];
  for (int i = 0; i < 4; ++i) o[i] = zacc;
  float Mrow[8], Lrow[8];
  for (int v = 0; v < 8; ++v) { Mrow[v] = -1e30f; Lrow[v] = 0.f; }

  const float scale = 0.10206207261596575f;  // 1/sqrt(96)
  const int njt = (m0b >> 5) + 2;            // causal prefix of 32-wide key tiles

  // Prologue: TDM K tile j=0 into buffer 0.
  if (wid == 0) {
    tdm_load_2d_bf16(lds_offset_of(&Kt[0][0]),
                     kf + headbase * 96, 96u, 32u, 96u);
  }

  for (int j = 0; j < njt; ++j) {
    const int buf = j & 1;
    if (wid == 0) {
      if (j + 1 < njt) {
        tdm_load_2d_bf16(lds_offset_of(&Kt[buf ^ 1][0]),
                         kf + (headbase + (size_t)(j + 1) * 32) * 96,
                         96u, 32u, 96u);
        __builtin_amdgcn_s_wait_tensorcnt(1);
      } else {
        __builtin_amdgcn_s_wait_tensorcnt(0);
      }
    }
    // V tile transposed: 256 chunks across 128 threads
    for (int c = tid; c < 256; c += 128) {
      int row = c >> 3, col8 = (c & 7) * 8;
      u32x4 d = *(const u32x4*)(vT + (headbase + j * 32 + row) * 64 + col8);
      const unsigned short* p = (const unsigned short*)&d;
      for (int i = 0; i < 8; ++i) Vt[(col8 + i) * 32 + row] = p[i];
    }
    // Prefetch next V tile rows into cache while this tile is consumed.
    if (j + 1 < njt && tid < 32)
      __builtin_prefetch(vT + (headbase + (size_t)(j + 1) * 32 + tid) * 64, 0, 1);
    __syncthreads();

    if (j * 32 <= m0w + 15) {                 // causal: wave needs this tile
      const unsigned short* Kb = &Kt[buf][0];
      f32x8 s0 = zacc, s1 = zacc;
      const int kb = (lane >> 4) * 8;
      const int r0 = lane & 15, r1 = 16 + (lane & 15);
      // Load all six K fragments, then burst six WMMAs.
      FragB16 bk[6];
      for (int kc = 0; kc < 3; ++kc) {
        bk[kc * 2 + 0].q[0] = *(const u32x4*)(Kb + r0 * 96 + kc * 32 + kb);
        bk[kc * 2 + 0].q[1] = *(const u32x4*)(Kb + r0 * 96 + kc * 32 + kb + 16);
        bk[kc * 2 + 1].q[0] = *(const u32x4*)(Kb + r1 * 96 + kc * 32 + kb);
        bk[kc * 2 + 1].q[1] = *(const u32x4*)(Kb + r1 * 96 + kc * 32 + kb + 16);
      }
      for (int kc = 0; kc < 3; ++kc) {
        s0 = wmma_bf16(aq[kc], bk[kc * 2 + 0], s0);
        s1 = wmma_bf16(aq[kc], bk[kc * 2 + 1], s1);
      }
      const int col0 = j * 32 + (lane & 15);
      const int col1 = col0 + 16;
      const int halfoff = (lane >> 4) * 8;
      for (int v = 0; v < 8; ++v) {
        int qi = m0w + v + halfoff;
        float e0 = (col0 <= qi) ? s0[v] * scale : -1e30f;
        float e1 = (col1 <= qi) ? s1[v] * scale : -1e30f;
        float tmax = fmaxf(e0, e1);
        for (int off = 8; off >= 1; off >>= 1)
          tmax = fmaxf(tmax, __shfl_xor(tmax, off, 16));
        float mnew = fmaxf(Mrow[v], tmax);
        float corr = __expf(Mrow[v] - mnew);
        float p0 = __expf(e0 - mnew), p1 = __expf(e1 - mnew);
        float rsum = p0 + p1;
        for (int off = 8; off >= 1; off >>= 1)
          rsum += __shfl_xor(rsum, off, 16);
        Lrow[v] = Lrow[v] * corr + rsum;
        Mrow[v] = mnew;
        for (int nt = 0; nt < 4; ++nt) o[nt][v] *= corr;
        int mrow = v + halfoff;
        Pt[wid][mrow * 32 + (lane & 15)]      = f2bf(p0);
        Pt[wid][mrow * 32 + 16 + (lane & 15)] = f2bf(p1);
      }
      asm volatile("s_wait_dscnt 0" ::: "memory");  // order P write -> P read (same wave)
      FragB16 pa;
      {
        int row = lane & 15, c0 = (lane >> 4) * 8;
        pa.q[0] = *(const u32x4*)(Pt[wid] + row * 32 + c0);
        pa.q[1] = *(const u32x4*)(Pt[wid] + row * 32 + c0 + 16);
      }
      FragB16 bv[4];
      for (int nt = 0; nt < 4; ++nt) {
        int col = nt * 16 + (lane & 15);
        bv[nt].q[0] = *(const u32x4*)(Vt + col * 32 + kb);
        bv[nt].q[1] = *(const u32x4*)(Vt + col * 32 + kb + 16);
      }
      for (int nt = 0; nt < 4; ++nt) o[nt] = wmma_bf16(pa, bv[nt], o[nt]);
    }
    __syncthreads();
  }

  // normalize + store bf16 into [b*t, 1024], columns h*64 + n
  for (int v = 0; v < 8; ++v) {
    float inv = 1.0f / Lrow[v];
    int mrow = m0w + v + (lane >> 4) * 8;
    size_t rowbase = ((size_t)bi * T + mrow) * 1024 + h * 64;
    for (int nt = 0; nt < 4; ++nt)
      attn[rowbase + nt * 16 + (lane & 15)] = f2bf(o[nt][v] * inv);
  }
}

// ---------------------------------------------------------------------------
// Host orchestration
// ---------------------------------------------------------------------------
extern "C" void kernel_launch(void* const* d_in, const int* in_sizes, int n_in,
                              void* d_out, int out_size, void* d_ws, size_t ws_size,
                              hipStream_t stream) {
  const int B = 2, T = 2048, DM = 1024, NH = 16, DC1 = 384, DC = 256, DR = 32;
  const int M = B * T;  // 4096

  const float* x     = (const float*)d_in[0];
  const float* w_dq  = (const float*)d_in[1];
  const float* w_uq  = (const float*)d_in[2];
  const float* w_rq  = (const float*)d_in[3];
  const float* w_dkv = (const float*)d_in[4];
  const float* w_rk  = (const float*)d_in[5];
  const float* w_uk  = (const float*)d_in[6];
  const float* w_uv  = (const float*)d_in[7];
  const float* w_o   = (const float*)d_in[8];
  float* out = (float*)d_out;

  char* ws = (char*)d_ws;
  size_t off = 0;
  auto alloc = [&](size_t bytes) -> char* {
    char* p = ws + off;
    off += (bytes + 255) & ~(size_t)255;
    return p;
  };
  typedef unsigned short us;
  us* xb    = (us*)alloc((size_t)M * DM * 2);
  us* wdqb  = (us*)alloc((size_t)DM * DC1 * 2);
  us* wuqb  = (us*)alloc((size_t)DC1 * DM * 2);
  us* wrqb  = (us*)alloc((size_t)DC1 * NH * DR * 2);
  us* wdkvb = (us*)alloc((size_t)DM * DC * 2);
  us* wrkb  = (us*)alloc((size_t)DM * DR * 2);
  us* wukb  = (us*)alloc((size_t)DC * DM * 2);
  us* wuvb  = (us*)alloc((size_t)DC * DM * 2);
  us* wob   = (us*)alloc((size_t)DM * DM * 2);
  us* cqb   = (us*)alloc((size_t)M * DC1 * 2);
  us* ckvb  = (us*)alloc((size_t)M * DC * 2);
  float* krotf   = (float*)alloc((size_t)M * DR * 4);
  float* qstatef = (float*)alloc((size_t)M * DM * 4);
  float* qrotf   = (float*)alloc((size_t)M * NH * DR * 4);
  float* kstatef = (float*)alloc((size_t)M * DM * 4);
  float* vstatef = (float*)alloc((size_t)M * DM * 4);
  us* qfb   = (us*)alloc((size_t)B * NH * T * 96 * 2);
  us* kfb   = (us*)alloc((size_t)B * NH * T * 96 * 2);
  us* vTb   = (us*)alloc((size_t)B * NH * T * 64 * 2);
  us* attnb = (us*)alloc((size_t)M * DM * 2);
  (void)ws_size; (void)in_sizes; (void)n_in; (void)out_size;

  auto conv = [&](const float* s, us* d, int n) {
    k_f32_to_bf16<<<512, 256, 0, stream>>>(s, d, n);
  };
  conv(x,     xb,    M * DM);
  conv(w_dq,  wdqb,  DM * DC1);
  conv(w_uq,  wuqb,  DC1 * DM);
  conv(w_rq,  wrqb,  DC1 * NH * DR);
  conv(w_dkv, wdkvb, DM * DC);
  conv(w_rk,  wrkb,  DM * DR);
  conv(w_uk,  wukb,  DC * DM);
  conv(w_uv,  wuvb,  DC * DM);
  conv(w_o,   wob,   DM * DM);

  auto gemm = [&](const us* A, const us* Bm, float* Cf, us* Cb,
                  int m, int n, int k) {
    dim3 grid((n + 127) / 128, m / 64);
    k_gemm_bf16<<<grid, 256, 0, stream>>>(A, Bm, Cf, Cb, m, n, k);
  };
  gemm(xb,   wdqb,  nullptr, cqb,  M, DC1, DM);         // c_q
  gemm(xb,   wdkvb, nullptr, ckvb, M, DC,  DM);         // c_kv
  gemm(xb,   wrkb,  krotf,   nullptr, M, DR, DM);       // k_rot pre
  gemm(cqb,  wuqb,  qstatef, nullptr, M, DM, DC1);      // q_state
  gemm(cqb,  wrqb,  qrotf,   nullptr, M, NH * DR, DC1); // q_rot pre
  gemm(ckvb, wukb,  kstatef, nullptr, M, DM, DC);       // k_state
  gemm(ckvb, wuvb,  vstatef, nullptr, M, DM, DC);       // v_state

  k_prep<<<M, 256, 0, stream>>>(qstatef, qrotf, kstatef, krotf, vstatef,
                                qfb, kfb, vTb, T);

  dim3 agrid(T / 64, NH, B);
  k_attn<<<agrid, 128, 0, stream>>>(qfb, kfb, vTb, attnb, T);

  gemm(attnb, wob, out, nullptr, M, DM, DM);            // out = attn @ w_o
}
